// SelectiveSSM_59828894433547
// MI455X (gfx1250) — matrix operand order repaired and checked by reading, not compiled, not run on hardware
//
#include <hip/hip_runtime.h>

#define D_MODEL 1024
#define D_STATE 16
#define BATCH   4
#define SEQ     2048
#define MTOT    (BATCH*SEQ)      /* 8192 */
#define LCHUNK  128
#define NCHUNK  (SEQ/LCHUNK)     /* 16 */

typedef __bf16  v8bf  __attribute__((ext_vector_type(8)));
typedef __bf16  v16bf __attribute__((ext_vector_type(16)));
typedef float   v8f   __attribute__((ext_vector_type(8)));
typedef int     v4i   __attribute__((ext_vector_type(4)));

#define BM 128
#define BN 64
#define BK 32
#define XS 40   /* padded LDS row stride (bf16): 80B = 20 dwords -> conflict-free, 16B aligned */

/* ---- gfx1250 async global->LDS path (guarded; falls back to reg round-trip) ----
 * builtin signature (from hipcc diagnostics): (v4i AS1*, v4i AS3*, imm i32 offset, imm i32 cpol) */
#if __has_builtin(__builtin_amdgcn_global_load_async_to_lds_b128) && \
    __has_builtin(__builtin_amdgcn_s_wait_asynccnt)
#define HAVE_ASYNC_LDS 1
#define ASYNC_CP16(gsrc, ldst)                                              \
  __builtin_amdgcn_global_load_async_to_lds_b128(                           \
      (__attribute__((address_space(1))) v4i*)(gsrc),                       \
      (__attribute__((address_space(3))) v4i*)(ldst), 0, 0)
#define ASYNC_WAIT() __builtin_amdgcn_s_wait_asynccnt(0)
#else
#define HAVE_ASYNC_LDS 0
#endif

__device__ __forceinline__ unsigned short f2bf(float f) {
  unsigned u = __float_as_uint(f);
  unsigned r = u + 0x7FFFu + ((u >> 16) & 1u);   /* round-to-nearest-even */
  return (unsigned short)(r >> 16);
}
__device__ __forceinline__ float bf2f(unsigned short h) {
  return __uint_as_float(((unsigned)h) << 16);
}

/* A fragment (16x32 bf16): lane<16 -> M=lane, K={0..7,16..23}; lane>=16 -> M=lane-16, K={8..15,24..31} */
__device__ __forceinline__ v16bf load_fragA(const unsigned short* base, int lane) {
  int row = lane & 15;
  int kb  = (lane >> 4) * 8;
  const v8bf* p0 = (const v8bf*)(base + row * XS + kb);
  const v8bf* p1 = (const v8bf*)(base + row * XS + kb + 16);
  union { v16bf v; v8bf h[2]; } u;
  u.h[0] = *p0; u.h[1] = *p1;
  return u.v;
}
/* B fragment (32x16 bf16): lane<16 -> N=lane, K={0..7,16..23}; lane>=16 -> N=lane-16, K={8..15,24..31} */
__device__ __forceinline__ v16bf load_fragB(const unsigned short* base, int lane) {
  int row = lane & 15;
  int kb  = (lane >> 4) * 8;
  const v8bf* p0 = (const v8bf*)(base + row * XS + kb);
  const v8bf* p1 = (const v8bf*)(base + row * XS + kb + 16);
  union { v16bf v; v8bf h[2]; } u;
  u.h[0] = *p0; u.h[1] = *p1;
  return u.v;
}

#define WMMA_BF16(A,B,C) \
  __builtin_amdgcn_wmma_f32_16x16x32_bf16(false,(A),false,(B),(short)0,(C),false,false)

/* split-bf16 3-term product: 12 WMMAs over a 32x32 wave tile */
__device__ __forceinline__ void wmma_block(const unsigned short* aH, const unsigned short* aL,
                                           const unsigned short* bH, const unsigned short* bL,
                                           int wm, int wn, int lane, v8f (&acc)[2][2])
{
  v16bf fAh[2], fAl[2], fBh[2], fBl[2];
  #pragma unroll
  for (int i = 0; i < 2; ++i) {
    fAh[i] = load_fragA(aH + (wm*32 + i*16)*XS, lane);
    fAl[i] = load_fragA(aL + (wm*32 + i*16)*XS, lane);
  }
  #pragma unroll
  for (int j = 0; j < 2; ++j) {
    fBh[j] = load_fragB(bH + (wn*32 + j*16)*XS, lane);
    fBl[j] = load_fragB(bL + (wn*32 + j*16)*XS, lane);
  }
  #pragma unroll
  for (int i = 0; i < 2; ++i)
    #pragma unroll
    for (int j = 0; j < 2; ++j) {
      acc[i][j] = WMMA_BF16(fAh[i], fBh[j], acc[i][j]);
      acc[i][j] = WMMA_BF16(fAh[i], fBl[j], acc[i][j]);
      acc[i][j] = WMMA_BF16(fAl[i], fBh[j], acc[i][j]);
    }
}

/* ---------------- Kernel 0: one-shot fp32 -> (bf16 hi, bf16 lo) split ---------------- */
__global__ __launch_bounds__(256)
void cvt_split_bf16(const float* __restrict__ src, unsigned short* __restrict__ hi,
                    unsigned short* __restrict__ lo, int n)
{
  int i = (blockIdx.x * 256 + threadIdx.x) * 4;
  if (i >= n) return;
  float4 v = *(const float4*)(src + i);
  const float* f = (const float*)&v;
  unsigned short h[4], l[4];
  #pragma unroll
  for (int q = 0; q < 4; ++q) {
    h[q] = f2bf(f[q]);
    l[q] = f2bf(f[q] - bf2f(h[q]));
  }
  uint2 hp, lp;
  hp.x = (unsigned)h[0] | ((unsigned)h[1] << 16);
  hp.y = (unsigned)h[2] | ((unsigned)h[3] << 16);
  lp.x = (unsigned)l[0] | ((unsigned)l[1] << 16);
  lp.y = (unsigned)l[2] | ((unsigned)l[3] << 16);
  *(uint2*)(hi + i) = hp;
  *(uint2*)(lo + i) = lp;
}

/* ---------------- Kernel A: dt = softplus(x @ W_dt^T + b_dt) ---------------- */
__global__ __launch_bounds__(256)
void ssm_gemm_dt(const unsigned short* __restrict__ xh, const unsigned short* __restrict__ xl,
                 const unsigned short* __restrict__ wh, const unsigned short* __restrict__ wl,
                 const float* __restrict__ bias, float* __restrict__ dt)
{
  __shared__ unsigned short aH[2][BM*XS], aL[2][BM*XS];
  __shared__ unsigned short bH[2][BN*XS], bL[2][BN*XS];

  const int tid  = threadIdx.x;
  const int bm   = blockIdx.x;          /* M / 128 */
  const int bn   = blockIdx.y;          /* N / 64  */
  const int wave = tid >> 5, lane = tid & 31;
  const int wm   = wave >> 1, wn = wave & 1;

  v8f acc[2][2];
  #pragma unroll
  for (int i = 0; i < 2; ++i)
    #pragma unroll
    for (int j = 0; j < 2; ++j) acc[i][j] = 0.0f;

  const int arow = tid >> 1, acol = (tid & 1) * 16;   /* A: 128 rows x 32 cols, 16 bf16/thread */
  const int brow = tid >> 2, bcol = (tid & 3) * 8;    /* B:  64 rows x 32 cols,  8 bf16/thread */
  const unsigned short* axh = xh + (size_t)(bm*BM + arow) * D_MODEL + acol;
  const unsigned short* axl = xl + (size_t)(bm*BM + arow) * D_MODEL + acol;
  const unsigned short* bwh = wh + (size_t)(bn*BN + brow) * D_MODEL + bcol;
  const unsigned short* bwl = wl + (size_t)(bn*BN + brow) * D_MODEL + bcol;
  const unsigned sa = arow*XS + acol;
  const unsigned sb = brow*XS + bcol;
  const int NK = D_MODEL / BK;

#if HAVE_ASYNC_LDS
  /* async double-buffer: 1 barrier/iter, copies overlap WMMA */
  #define STAGE_ASYNC(s, k0) do {                      \
    ASYNC_CP16(axh + (k0),     &aH[s][sa]);            \
    ASYNC_CP16(axh + (k0) + 8, &aH[s][sa + 8]);        \
    ASYNC_CP16(axl + (k0),     &aL[s][sa]);            \
    ASYNC_CP16(axl + (k0) + 8, &aL[s][sa + 8]);        \
    ASYNC_CP16(bwh + (k0),     &bH[s][sb]);            \
    ASYNC_CP16(bwl + (k0),     &bL[s][sb]);            \
  } while (0)
  STAGE_ASYNC(0, 0);
  for (int kt = 0; kt < NK; ++kt) {
    ASYNC_WAIT();
    __syncthreads();
    if (kt + 1 < NK) STAGE_ASYNC((kt + 1) & 1, (kt + 1) * BK);
    wmma_block(aH[kt & 1], aL[kt & 1], bH[kt & 1], bL[kt & 1], wm, wn, lane, acc);
  }
  #undef STAGE_ASYNC
#else
  /* register double-buffer: loads for k+1 issue before compute(k) */
  uint4 r0, r1, r2, r3, r4, r5;
  #define LOADR(k0) do {                          \
    r0 = *(const uint4*)(axh + (k0));             \
    r1 = *(const uint4*)(axh + (k0) + 8);         \
    r2 = *(const uint4*)(axl + (k0));             \
    r3 = *(const uint4*)(axl + (k0) + 8);         \
    r4 = *(const uint4*)(bwh + (k0));             \
    r5 = *(const uint4*)(bwl + (k0));             \
  } while (0)
  #define STORER(s) do {                          \
    *(uint4*)&aH[s][sa]     = r0;                 \
    *(uint4*)&aH[s][sa + 8] = r1;                 \
    *(uint4*)&aL[s][sa]     = r2;                 \
    *(uint4*)&aL[s][sa + 8] = r3;                 \
    *(uint4*)&bH[s][sb]     = r4;                 \
    *(uint4*)&bL[s][sb]     = r5;                 \
  } while (0)
  LOADR(0); STORER(0);
  __syncthreads();
  for (int kt = 0; kt < NK; ++kt) {
    if (kt + 1 < NK) LOADR((kt + 1) * BK);
    wmma_block(aH[kt & 1], aL[kt & 1], bH[kt & 1], bL[kt & 1], wm, wn, lane, acc);
    __syncthreads();
    if (kt + 1 < NK) STORER((kt + 1) & 1);
    __syncthreads();
  }
  #undef LOADR
  #undef STORER
#endif

  /* epilogue: bias + fast softplus.  D layout: vgpr v -> M=(lane>=16?8:0)+v, N=lane&15 */
  const int n = lane & 15, mt = (lane >> 4) * 8;
  #pragma unroll
  for (int i = 0; i < 2; ++i)
    #pragma unroll
    for (int j = 0; j < 2; ++j) {
      const int col = bn*BN + wn*32 + j*16 + n;
      const float bv = bias[col];
      const size_t rb = (size_t)(bm*BM + wm*32 + i*16 + mt);
      #pragma unroll
      for (int v = 0; v < 8; ++v) {
        float z  = acc[i][j][v] + bv;
        float sp = fmaxf(z, 0.0f) + __logf(1.0f + __expf(-fabsf(z)));
        dt[(rb + v) * (size_t)D_MODEL + col] = sp;
      }
    }
}

/* ---------------- Kernel A2: Bp = x@W_B^T, Cp = x@W_C^T (N-tile 16+16) ---------------- */
__global__ __launch_bounds__(256)
void ssm_gemm_bc(const unsigned short* __restrict__ xh, const unsigned short* __restrict__ xl,
                 const unsigned short* __restrict__ wbh, const unsigned short* __restrict__ wbl,
                 const unsigned short* __restrict__ wch, const unsigned short* __restrict__ wcl,
                 float* __restrict__ Bp, float* __restrict__ Cp)
{
  __shared__ unsigned short aHs[BM*XS], aLs[BM*XS];
  __shared__ unsigned short bHs[32*XS], bLs[32*XS];

  const int tid  = threadIdx.x;
  const int bm   = blockIdx.x;
  const int wave = tid >> 5, lane = tid & 31;
  const int wm   = wave >> 1;     /* 0..3 */
  const int wsel = wave & 1;      /* 0 -> Bp, 1 -> Cp */

  v8f acc[2];
  acc[0] = 0.0f; acc[1] = 0.0f;

  const int arow = tid >> 1, acol = (tid & 1) * 16;
  const int brow = tid >> 3;           /* 0..31: rows 0-15 = W_B, 16-31 = W_C */
  const int bcol = (tid & 7) * 4;
  const unsigned short* axh = xh + (size_t)(bm*BM + arow) * D_MODEL + acol;
  const unsigned short* axl = xl + (size_t)(bm*BM + arow) * D_MODEL + acol;
  const unsigned short* bwh = (brow < 16 ? wbh + (size_t)brow * D_MODEL
                                         : wch + (size_t)(brow - 16) * D_MODEL) + bcol;
  const unsigned short* bwl = (brow < 16 ? wbl + (size_t)brow * D_MODEL
                                         : wcl + (size_t)(brow - 16) * D_MODEL) + bcol;
  const unsigned sa = arow*XS + acol;
  const unsigned sb = brow*XS + bcol;

  for (int k0 = 0; k0 < D_MODEL; k0 += BK) {
    uint4 r0 = *(const uint4*)(axh + k0);
    uint4 r1 = *(const uint4*)(axh + k0 + 8);
    uint4 r2 = *(const uint4*)(axl + k0);
    uint4 r3 = *(const uint4*)(axl + k0 + 8);
    uint2 w0 = *(const uint2*)(bwh + k0);
    uint2 w1 = *(const uint2*)(bwl + k0);
    __syncthreads();
    *(uint4*)&aHs[sa]     = r0;
    *(uint4*)&aHs[sa + 8] = r1;
    *(uint4*)&aLs[sa]     = r2;
    *(uint4*)&aLs[sa + 8] = r3;
    *(uint2*)&bHs[sb]     = w0;
    *(uint2*)&bLs[sb]     = w1;
    __syncthreads();

    v16bf fBh = load_fragB(bHs + (wsel*16)*XS, lane);
    v16bf fBl = load_fragB(bLs + (wsel*16)*XS, lane);
    #pragma unroll
    for (int i = 0; i < 2; ++i) {
      v16bf fAh = load_fragA(aHs + (wm*32 + i*16)*XS, lane);
      v16bf fAl = load_fragA(aLs + (wm*32 + i*16)*XS, lane);
      acc[i] = WMMA_BF16(fAh, fBh, acc[i]);
      acc[i] = WMMA_BF16(fAh, fBl, acc[i]);
      acc[i] = WMMA_BF16(fAl, fBh, acc[i]);
    }
  }

  float* out = wsel ? Cp : Bp;
  const int n = lane & 15, mt = (lane >> 4) * 8;
  #pragma unroll
  for (int i = 0; i < 2; ++i) {
    const size_t rb = (size_t)(bm*BM + wm*32 + i*16 + mt);
    #pragma unroll
    for (int v = 0; v < 8; ++v)
      out[(rb + v) * (size_t)D_STATE + n] = acc[i][v];
  }
}

/* ---------------- Kernel B1: per-chunk local scan -> P (decay), S (final state) ----------------
 * P_n = prod_t exp(dt_t*A_n) = exp(A_n * sum_t dt_t): one exp per n per chunk. */
__global__ __launch_bounds__(256)
void ssm_scan_phase1(const float* __restrict__ dt, const float* __restrict__ x,
                     const float* __restrict__ Bp, const float* __restrict__ A_log,
                     float* __restrict__ Pst, float* __restrict__ Sst)
{
  const int d = blockIdx.x * 256 + threadIdx.x;  /* 0..1023 */
  const int c = blockIdx.y;
  const int b = blockIdx.z;
  const int t0 = c * LCHUNK;

  __shared__ float sB[LCHUNK * D_STATE];
  const float* bpb = Bp + ((size_t)b * SEQ + t0) * D_STATE;
#if HAVE_ASYNC_LDS
  ASYNC_CP16(bpb + threadIdx.x * 4,        &sB[threadIdx.x * 4]);
  ASYNC_CP16(bpb + threadIdx.x * 4 + 1024, &sB[threadIdx.x * 4 + 1024]);
  ASYNC_WAIT();
#else
  for (int i = threadIdx.x; i < LCHUNK * D_STATE; i += 256) sB[i] = bpb[i];
#endif
  __syncthreads();

  float An[D_STATE];
  #pragma unroll
  for (int nn = 0; nn < D_STATE; ++nn) An[nn] = -__expf(A_log[(size_t)d * D_STATE + nn]);

  float h[D_STATE];
  #pragma unroll
  for (int nn = 0; nn < D_STATE; ++nn) h[nn] = 0.0f;
  float dts = 0.0f;

  const float* dtp = dt + ((size_t)b * SEQ + t0) * D_MODEL + d;
  const float* xp  = x  + ((size_t)b * SEQ + t0) * D_MODEL + d;
  for (int t = 0; t < LCHUNK; ++t) {
    const float e  = dtp[(size_t)t * D_MODEL];
    const float xv = xp [(size_t)t * D_MODEL];
    dts += e;
    const float exv = e * xv;
    const float* bt = &sB[t * D_STATE];
    #pragma unroll
    for (int nn = 0; nn < D_STATE; ++nn) {
      float dA = __expf(e * An[nn]);
      h[nn] = fmaf(dA, h[nn], exv * bt[nn]);
    }
  }

  const size_t base = (((size_t)b * NCHUNK + c) * D_MODEL + d) * D_STATE;
  #pragma unroll
  for (int nn = 0; nn < D_STATE; ++nn) {
    Pst[base + nn] = __expf(An[nn] * dts);
    Sst[base + nn] = h[nn];
  }
}

/* ---------------- Kernel B2: sequential combine over chunks ---------------- */
__global__ __launch_bounds__(256)
void ssm_scan_combine(const float* __restrict__ Pst, const float* __restrict__ Sst,
                      float* __restrict__ Hinit)
{
  const int idx = blockIdx.x * 256 + threadIdx.x;   /* 65536 = B*D*N */
  const int nn = idx & 15;
  const int d  = (idx >> 4) & (D_MODEL - 1);
  const int b  = idx >> 14;
  float hc = 0.0f;
  for (int c = 0; c < NCHUNK; ++c) {
    const size_t off = (((size_t)b * NCHUNK + c) * D_MODEL + d) * D_STATE + nn;
    Hinit[off] = hc;
    hc = fmaf(Pst[off], hc, Sst[off]);
  }
}

/* ---------------- Kernel B3: re-scan chunks with correct init state, emit y ---------------- */
__global__ __launch_bounds__(256)
void ssm_scan_phase2(const float* __restrict__ dt, const float* __restrict__ x,
                     const float* __restrict__ Bp, const float* __restrict__ Cp,
                     const float* __restrict__ A_log, const float* __restrict__ Dp,
                     const float* __restrict__ Hinit, float* __restrict__ y)
{
  const int d = blockIdx.x * 256 + threadIdx.x;
  const int c = blockIdx.y;
  const int b = blockIdx.z;
  const int t0 = c * LCHUNK;

  __shared__ float sB[LCHUNK * D_STATE];
  __shared__ float sC[LCHUNK * D_STATE];
  const float* bpb = Bp + ((size_t)b * SEQ + t0) * D_STATE;
  const float* cpb = Cp + ((size_t)b * SEQ + t0) * D_STATE;
#if HAVE_ASYNC_LDS
  ASYNC_CP16(bpb + threadIdx.x * 4,        &sB[threadIdx.x * 4]);
  ASYNC_CP16(bpb + threadIdx.x * 4 + 1024, &sB[threadIdx.x * 4 + 1024]);
  ASYNC_CP16(cpb + threadIdx.x * 4,        &sC[threadIdx.x * 4]);
  ASYNC_CP16(cpb + threadIdx.x * 4 + 1024, &sC[threadIdx.x * 4 + 1024]);
  ASYNC_WAIT();
#else
  for (int i = threadIdx.x; i < LCHUNK * D_STATE; i += 256) { sB[i] = bpb[i]; sC[i] = cpb[i]; }
#endif
  __syncthreads();

  float An[D_STATE];
  #pragma unroll
  for (int nn = 0; nn < D_STATE; ++nn) An[nn] = -__expf(A_log[(size_t)d * D_STATE + nn]);

  float h[D_STATE];
  const size_t hbase = (((size_t)b * NCHUNK + c) * D_MODEL + d) * D_STATE;
  #pragma unroll
  for (int nn = 0; nn < D_STATE; ++nn) h[nn] = Hinit[hbase + nn];

  const float Dv = Dp[d];
  const float* dtp = dt + ((size_t)b * SEQ + t0) * D_MODEL + d;
  const float* xp  = x  + ((size_t)b * SEQ + t0) * D_MODEL + d;
  float* yp        = y  + ((size_t)b * SEQ + t0) * D_MODEL + d;

  for (int t = 0; t < LCHUNK; ++t) {
    const float e  = dtp[(size_t)t * D_MODEL];
    const float xv = xp [(size_t)t * D_MODEL];
    const float exv = e * xv;
    const float* bt = &sB[t * D_STATE];
    const float* ct = &sC[t * D_STATE];
    float acc = 0.0f;
    #pragma unroll
    for (int nn = 0; nn < D_STATE; ++nn) {
      float dA = __expf(e * An[nn]);
      h[nn] = fmaf(dA, h[nn], exv * bt[nn]);
      acc   = fmaf(h[nn], ct[nn], acc);
    }
    yp[(size_t)t * D_MODEL] = fmaf(Dv, xv, acc);
  }
}

/* ---------------- Host launcher ---------------- */
extern "C" void kernel_launch(void* const* d_in, const int* in_sizes, int n_in,
                              void* d_out, int out_size, void* d_ws, size_t ws_size,
                              hipStream_t stream) {
  (void)in_sizes; (void)n_in; (void)out_size; (void)ws_size;
  const float* x     = (const float*)d_in[0];
  const float* W_dt  = (const float*)d_in[1];
  const float* b_dt  = (const float*)d_in[2];
  const float* W_B   = (const float*)d_in[3];
  const float* W_C   = (const float*)d_in[4];
  const float* A_log = (const float*)d_in[5];
  const float* Dp    = (const float*)d_in[6];
  float* y = (float*)d_out;

  char* ws = (char*)d_ws;
  float* dt_ws = (float*)ws; ws += (size_t)MTOT * D_MODEL * sizeof(float);         /* 33.5 MB */
  float* Bp    = (float*)ws; ws += (size_t)MTOT * D_STATE * sizeof(float);
  float* Cp    = (float*)ws; ws += (size_t)MTOT * D_STATE * sizeof(float);
  float* Pst   = (float*)ws; ws += (size_t)BATCH * NCHUNK * D_MODEL * D_STATE * sizeof(float);
  float* Sst   = (float*)ws; ws += (size_t)BATCH * NCHUNK * D_MODEL * D_STATE * sizeof(float);
  float* Hinit = (float*)ws; ws += (size_t)BATCH * NCHUNK * D_MODEL * D_STATE * sizeof(float);
  unsigned short* xh  = (unsigned short*)ws; ws += (size_t)MTOT * D_MODEL * 2;     /* 16 MB */
  unsigned short* xl  = (unsigned short*)ws; ws += (size_t)MTOT * D_MODEL * 2;
  unsigned short* wh  = (unsigned short*)ws; ws += (size_t)D_MODEL * D_MODEL * 2;  /* 2 MB */
  unsigned short* wl  = (unsigned short*)ws; ws += (size_t)D_MODEL * D_MODEL * 2;
  unsigned short* wbh = (unsigned short*)ws; ws += (size_t)D_STATE * D_MODEL * 2;  /* 32 KB */
  unsigned short* wbl = (unsigned short*)ws; ws += (size_t)D_STATE * D_MODEL * 2;
  unsigned short* wch = (unsigned short*)ws; ws += (size_t)D_STATE * D_MODEL * 2;
  unsigned short* wcl = (unsigned short*)ws;

  const int nx = MTOT * D_MODEL, nw = D_MODEL * D_MODEL, nbc = D_STATE * D_MODEL;
  cvt_split_bf16<<<dim3(nx / 1024),  256, 0, stream>>>(x,    xh,  xl,  nx);
  cvt_split_bf16<<<dim3(nw / 1024),  256, 0, stream>>>(W_dt, wh,  wl,  nw);
  cvt_split_bf16<<<dim3(nbc / 1024), 256, 0, stream>>>(W_B,  wbh, wbl, nbc);
  cvt_split_bf16<<<dim3(nbc / 1024), 256, 0, stream>>>(W_C,  wch, wcl, nbc);

  ssm_gemm_dt<<<dim3(MTOT/BM, D_MODEL/BN), 256, 0, stream>>>(xh, xl, wh, wl, b_dt, dt_ws);
  ssm_gemm_bc<<<dim3(MTOT/BM),             256, 0, stream>>>(xh, xl, wbh, wbl, wch, wcl, Bp, Cp);
  ssm_scan_phase1<<<dim3(D_MODEL/256, NCHUNK, BATCH), 256, 0, stream>>>(dt_ws, x, Bp, A_log, Pst, Sst);
  ssm_scan_combine<<<dim3((BATCH*D_MODEL*D_STATE)/256), 256, 0, stream>>>(Pst, Sst, Hinit);
  ssm_scan_phase2<<<dim3(D_MODEL/256, NCHUNK, BATCH), 256, 0, stream>>>(dt_ws, x, Bp, Cp, A_log, Dp, Hinit, y);
}
